// Qwen3NextLinearAttention_74457553043853
// MI455X (gfx1250) — matrix operand order repaired and checked by reading, not compiled, not run on hardware
//
#include <hip/hip_runtime.h>
#include <hip/hip_bf16.h>

typedef __bf16 bf16_t;
typedef __attribute__((ext_vector_type(16))) __bf16 v16bf;
typedef __attribute__((ext_vector_type(8)))  __bf16 v8bf;
typedef __attribute__((ext_vector_type(8)))  float  v8f;

#define HID   2048
#define NKH   16
#define NVH   32
#define DK    128
#define DV    128
#define TOK   4096            // b*s = 2*2048
#define SEQ   2048
#define QKVZ_DIM 12288
#define CONV_DIM 8192
#define KEY_DIM  2048
#define VAL_DIM  4096

// ---------------- workspace layout (bytes) ----------------
#define OFF_HB    ((size_t)0)                    // h bf16        [4096,2048]  16 MB
#define OFF_WQT   ((size_t)16777216)             // Wqkvz^T bf16  [12288,2048] 48 MB
#define OFF_QKVZ  ((size_t)67108864)             // qkvz bf16     [4096,12288] 96 MB
#define OFF_Q     ((size_t)167772160)            // q f32         [4096,16,128] 32 MB
#define OFF_K     ((size_t)201326592)            // k f32         32 MB
#define OFF_V     ((size_t)234881024)            // v f32         [4096,32,128] 64 MB
#define OFF_G     ((size_t)301989888)            // g f32         [4096,32]
#define OFF_BETA  ((size_t)302514176)            // beta f32
#define OFF_OG    ((size_t)303038464)            // gated out bf16 [4096,4096] 32 MB
#define OFF_WOT   ((size_t)336592896)            // Wout^T bf16   [2048,4096]  16 MB
#define OFF_O     ((size_t)0)                    // scan out f32 [4096,32,128] 64 MB (aliases HB+WQT, dead after GEMM1)

// ---------------- pack / convert ----------------
__global__ void pack_h_kernel(const float* __restrict__ hid, const float* __restrict__ mask,
                              bf16_t* __restrict__ hb) {
  int i = blockIdx.x * 256 + threadIdx.x;
  if (i < TOK * HID) {
    int t = i >> 11;
    hb[i] = (bf16_t)(hid[i] * mask[t]);
  }
}

__global__ void transpose_bf16_kernel(const float* __restrict__ W, bf16_t* __restrict__ Wt,
                                      int K, int N) {
  long i = (long)blockIdx.x * 256 + threadIdx.x;
  if (i < (long)K * N) {
    int n = (int)(i % N);
    int k = (int)(i / N);
    Wt[(long)n * K + k] = (bf16_t)W[i];
  }
}

// ---------------- WMMA fragment loaders (ISA 7.12.2 layouts) ----------------
// A 16x32 bf16: lane r=lane&15 holds row r; half h=lane>>4 holds K chunks
// [h*8, h*8+8) and [16+h*8, 16+h*8+8)   (interleaved per VGPR table)
__device__ __forceinline__ v16bf load_fragA(const bf16_t* __restrict__ p, int ld, int row0, int k0) {
  int lane = threadIdx.x & 31;
  int r = lane & 15, h = lane >> 4;
  const bf16_t* base = p + (long)(row0 + r) * ld + k0 + h * 8;
  v8bf lo = *(const v8bf*)(base);
  v8bf hi = *(const v8bf*)(base + 16);
  v16bf out;
#pragma unroll
  for (int i = 0; i < 8; ++i) { out[i] = lo[i]; out[i + 8] = hi[i]; }
  return out;
}

// B 32x16 bf16 (from B^T stored [N,K]): lane n=lane&15 holds column n;
// half h=lane>>4 holds K chunks [h*8, h*8+8) and [16+h*8, 16+h*8+8) (same as A)
__device__ __forceinline__ v16bf load_fragB(const bf16_t* __restrict__ p, int ld, int n0, int k0) {
  int lane = threadIdx.x & 31;
  int n = lane & 15, h = lane >> 4;
  const bf16_t* base = p + (long)(n0 + n) * ld + k0 + h * 8;
  v8bf lo = *(const v8bf*)(base);
  v8bf hi = *(const v8bf*)(base + 16);
  v16bf out;
#pragma unroll
  for (int i = 0; i < 8; ++i) { out[i] = lo[i]; out[i + 8] = hi[i]; }
  return out;
}

// ---------------- WMMA GEMM: C[M,N] = A[M,K] * B^T[N,K]^T ----------------
// block = 256 threads = 8 waves laid out 2(M) x 4(N); wave tile = 16 x 64.
template <bool BF16_OUT>
__global__ void wmma_gemm_kernel(const bf16_t* __restrict__ A, const bf16_t* __restrict__ Bt,
                                 void* __restrict__ Cp, int M, int N, int K) {
  int w  = threadIdx.x >> 5;
  int wm = w >> 2, wn = w & 3;
  int row0 = blockIdx.y * 32 + wm * 16;
  int n0   = blockIdx.x * 256 + wn * 64;

  v8f acc0 = {}, acc1 = {}, acc2 = {}, acc3 = {};

  for (int kb = 0; kb < K; kb += 32) {
    // speculative prefetch 4 K-slabs (128 elements) ahead (global_prefetch_b8)
    __builtin_prefetch(A + (long)row0 * K + kb + 128, 0, 1);
    __builtin_prefetch(Bt + (long)n0 * K + kb + 128, 0, 1);

    v16bf a  = load_fragA(A, K, row0, kb);
    v16bf b0 = load_fragB(Bt, K, n0 + 0,  kb);
    v16bf b1 = load_fragB(Bt, K, n0 + 16, kb);
    v16bf b2 = load_fragB(Bt, K, n0 + 32, kb);
    v16bf b3 = load_fragB(Bt, K, n0 + 48, kb);

    acc0 = __builtin_amdgcn_wmma_f32_16x16x32_bf16(false, a, false, b0, (short)0, acc0, false, false);
    acc1 = __builtin_amdgcn_wmma_f32_16x16x32_bf16(false, a, false, b1, (short)0, acc1, false, false);
    acc2 = __builtin_amdgcn_wmma_f32_16x16x32_bf16(false, a, false, b2, (short)0, acc2, false, false);
    acc3 = __builtin_amdgcn_wmma_f32_16x16x32_bf16(false, a, false, b3, (short)0, acc3, false, false);
  }

  // C/D layout: lane n = lane&15 holds column; VGPR r -> row (lane>>4)*8 + r
  int lane = threadIdx.x & 31;
  int n = lane & 15, mh = lane >> 4;
#pragma unroll
  for (int r = 0; r < 8; ++r) {
    long m = row0 + mh * 8 + r;
    if (BF16_OUT) {
      bf16_t* C = (bf16_t*)Cp;
      C[m * N + n0 + n +  0] = (bf16_t)acc0[r];
      C[m * N + n0 + n + 16] = (bf16_t)acc1[r];
      C[m * N + n0 + n + 32] = (bf16_t)acc2[r];
      C[m * N + n0 + n + 48] = (bf16_t)acc3[r];
    } else {
      float* C = (float*)Cp;
      C[m * N + n0 + n +  0] = acc0[r];
      C[m * N + n0 + n + 16] = acc1[r];
      C[m * N + n0 + n + 32] = acc2[r];
      C[m * N + n0 + n + 48] = acc3[r];
    }
  }
}

// ---------------- ba projection + gate math ----------------
__global__ void ba_gate_kernel(const float* __restrict__ hid, const float* __restrict__ mask,
                               const float* __restrict__ Wba, const float* __restrict__ A_log,
                               const float* __restrict__ dt_bias,
                               float* __restrict__ g, float* __restrict__ beta) {
  int i = blockIdx.x * 256 + threadIdx.x;
  if (i >= TOK * NVH) return;
  int t = i >> 5, vh = i & 31;
  int kh = vh >> 1, r = vh & 1;
  int cb = kh * 4 + r, ca = kh * 4 + 2 + r;
  const float* hrow = hid + (long)t * HID;
  float mk = mask[t];
  float sb = 0.f, sa = 0.f;
  for (int k = 0; k < HID; ++k) {
    float hv = hrow[k];
    sb = fmaf(hv, Wba[k * 64 + cb], sb);
    sa = fmaf(hv, Wba[k * 64 + ca], sa);
  }
  sb *= mk; sa *= mk;
  float x  = sa + dt_bias[vh];
  float sp = (x > 20.f) ? x : log1pf(__expf(x));       // softplus
  g[i]    = -__expf(A_log[vh]) * sp;
  beta[i] = 1.f / (1.f + __expf(-sb));                  // sigmoid
}

// ---------------- depthwise conv4 + SiLU + L2 norm ----------------
__global__ void conv_silu_norm_kernel(const bf16_t* __restrict__ qkvz, const float* __restrict__ ck,
                                      float* __restrict__ qout, float* __restrict__ kout,
                                      float* __restrict__ vout) {
  __shared__ float buf[CONV_DIM];
  int t = blockIdx.x;
  int s = t & (SEQ - 1);

  for (int c = threadIdx.x; c < CONV_DIM; c += 256) {
    int col;
    if (c < KEY_DIM) {                       // q channels
      int h = c >> 7; col = h * 768 + (c & 127);
    } else if (c < 2 * KEY_DIM) {            // k channels
      int cc = c - KEY_DIM; int h = cc >> 7; col = h * 768 + 128 + (cc & 127);
    } else {                                 // v channels
      int cc = c - 2 * KEY_DIM; int vh = cc >> 7; int kh = vh >> 1, rr = vh & 1;
      col = kh * 768 + 256 + rr * 128 + (cc & 127);
    }
    float acc = 0.f;
#pragma unroll
    for (int j = 0; j < 4; ++j) {
      int sj = s - 3 + j;
      if (sj >= 0)
        acc = fmaf((float)qkvz[(long)(t - 3 + j) * QKVZ_DIM + col], ck[j * CONV_DIM + c], acc);
    }
    buf[c] = acc / (1.f + __expf(-acc));     // SiLU
  }
  __syncthreads();

  int lane = threadIdx.x & 31, w = threadIdx.x >> 5;
  // pass 0: q (scale by DK^-0.5), pass 1: k
  for (int pass = 0; pass < 2; ++pass) {
    int   reg   = pass * KEY_DIM;
    float extra = (pass == 0) ? 0.08838834764831845f : 1.0f;   // 1/sqrt(128)
    float* outp = (pass == 0) ? qout : kout;
    for (int hh = w; hh < NKH; hh += 8) {
      int b0 = reg + hh * 128 + lane * 4;
      float x0 = buf[b0 + 0], x1 = buf[b0 + 1], x2 = buf[b0 + 2], x3 = buf[b0 + 3];
      float ss = x0 * x0 + x1 * x1 + x2 * x2 + x3 * x3;
#pragma unroll
      for (int off = 16; off > 0; off >>= 1) ss += __shfl_xor(ss, off, 32);
      float rs = rsqrtf(ss + 1e-6f) * extra;
      long ob = (long)t * KEY_DIM + hh * 128 + lane * 4;
      outp[ob + 0] = x0 * rs; outp[ob + 1] = x1 * rs;
      outp[ob + 2] = x2 * rs; outp[ob + 3] = x3 * rs;
    }
  }
  for (int i = threadIdx.x; i < VAL_DIM; i += 256)
    vout[(long)t * VAL_DIM + i] = buf[2 * KEY_DIM + i];
}

// ---------------- delta-rule recurrent scan ----------------
// one block per (batch, vhead); 128 threads; lane owns column v; S[:,v] in VGPRs
__global__ void __launch_bounds__(128)
scan_kernel(const float* __restrict__ q, const float* __restrict__ k, const float* __restrict__ v,
            const float* __restrict__ g, const float* __restrict__ beta, float* __restrict__ o) {
  const int bh = blockIdx.x;
  const int b  = bh >> 5, vh = bh & 31, kh = vh >> 1;
  const int tid = threadIdx.x;

  __shared__ float lk[8][DK];
  __shared__ float lq[8][DK];

  float S[DK];
#pragma unroll
  for (int i = 0; i < DK; ++i) S[i] = 0.f;

  for (int s0 = 0; s0 < SEQ; s0 += 8) {
    // stage q/k for the next 8 steps
    for (int i = tid; i < 8 * DK; i += 128) {
      int j = i >> 7, d = i & 127;
      long base = ((long)(b * SEQ + s0 + j) * NKH + kh) * DK + d;
      lk[j][d] = k[base];
      lq[j][d] = q[base];
    }
    __syncthreads();

    for (int j = 0; j < 8; ++j) {
      long ti = (long)(b * SEQ + s0 + j);
      float gg = g[ti * NVH + vh];
      float bb = beta[ti * NVH + vh];
      float vv = v[(ti * NVH + vh) * DV + tid];
      float eg = __expf(gg);

      float kv = 0.f;
#pragma unroll
      for (int kk = 0; kk < DK; ++kk) {
        S[kk] *= eg;
        kv = fmaf(lk[j][kk], S[kk], kv);
      }
      float delta = (vv - kv) * bb;
      float oo = 0.f;
#pragma unroll
      for (int kk = 0; kk < DK; ++kk) {
        S[kk] = fmaf(lk[j][kk], delta, S[kk]);
        oo = fmaf(lq[j][kk], S[kk], oo);
      }
      o[(ti * NVH + vh) * DV + tid] = oo;
    }
    __syncthreads();
  }
}

// ---------------- gated RMS norm * SiLU(z) -> bf16 ----------------
__global__ void gate_kernel(const float* __restrict__ o, const bf16_t* __restrict__ qkvz,
                            const float* __restrict__ normk, bf16_t* __restrict__ og) {
  int w = threadIdx.x >> 5, lane = threadIdx.x & 31;
  long row = (long)blockIdx.x * 8 + w;           // row over [TOK*NVH)
  int t  = (int)(row >> 5);
  int vh = (int)(row & 31);
  int kh = vh >> 1, r = vh & 1;
  int d0 = lane * 4;

  float x[4];
  float ss = 0.f;
#pragma unroll
  for (int i = 0; i < 4; ++i) {
    x[i] = o[row * DV + d0 + i];
    ss = fmaf(x[i], x[i], ss);
  }
#pragma unroll
  for (int off = 16; off > 0; off >>= 1) ss += __shfl_xor(ss, off, 32);
  float rs = rsqrtf(ss * (1.0f / DV) + 1e-6f);

#pragma unroll
  for (int i = 0; i < 4; ++i) {
    int d = d0 + i;
    float zz = (float)qkvz[(long)t * QKVZ_DIM + kh * 768 + 512 + r * 128 + d];
    float val = x[i] * rs * normk[d] * (zz / (1.f + __expf(-zz)));
    og[row * DV + d] = (bf16_t)val;
  }
}

// ---------------- launcher ----------------
extern "C" void kernel_launch(void* const* d_in, const int* in_sizes, int n_in,
                              void* d_out, int out_size, void* d_ws, size_t ws_size,
                              hipStream_t stream) {
  const float* hid   = (const float*)d_in[0];
  const float* mask  = (const float*)d_in[1];
  const float* Wqkvz = (const float*)d_in[2];
  const float* Wba   = (const float*)d_in[3];
  const float* ck    = (const float*)d_in[4];
  const float* A_log = (const float*)d_in[5];
  const float* dtb   = (const float*)d_in[6];
  const float* normk = (const float*)d_in[7];
  const float* Wout  = (const float*)d_in[8];

  char* ws = (char*)d_ws;
  bf16_t* hb    = (bf16_t*)(ws + OFF_HB);
  bf16_t* wqt   = (bf16_t*)(ws + OFF_WQT);
  bf16_t* qkvz  = (bf16_t*)(ws + OFF_QKVZ);
  float*  qbuf  = (float*)(ws + OFF_Q);
  float*  kbuf  = (float*)(ws + OFF_K);
  float*  vbuf  = (float*)(ws + OFF_V);
  float*  gbuf  = (float*)(ws + OFF_G);
  float*  bbuf  = (float*)(ws + OFF_BETA);
  bf16_t* og    = (bf16_t*)(ws + OFF_OG);
  bf16_t* wot   = (bf16_t*)(ws + OFF_WOT);
  float*  obuf  = (float*)(ws + OFF_O);      // aliases HB+WQT (dead after GEMM1)

  // 1) pack inputs to bf16
  pack_h_kernel<<<(TOK * HID) / 256, 256, 0, stream>>>(hid, mask, hb);
  transpose_bf16_kernel<<<(HID * QKVZ_DIM + 255) / 256, 256, 0, stream>>>(Wqkvz, wqt, HID, QKVZ_DIM);
  transpose_bf16_kernel<<<(VAL_DIM * HID + 255) / 256, 256, 0, stream>>>(Wout, wot, VAL_DIM, HID);

  // 2) qkvz = h @ W_qkvz  (M=4096, N=12288, K=2048) -> bf16
  wmma_gemm_kernel<true><<<dim3(QKVZ_DIM / 256, TOK / 32), 256, 0, stream>>>(
      hb, wqt, (void*)qkvz, TOK, QKVZ_DIM, HID);

  // 3) ba projection + g/beta
  ba_gate_kernel<<<(TOK * NVH) / 256, 256, 0, stream>>>(hid, mask, Wba, A_log, dtb, gbuf, bbuf);

  // 4) conv4 + SiLU + L2 norm -> q, k, v
  conv_silu_norm_kernel<<<TOK, 256, 0, stream>>>(qkvz, ck, qbuf, kbuf, vbuf);

  // 5) recurrent delta-rule scan
  scan_kernel<<<2 * NVH, 128, 0, stream>>>(qbuf, kbuf, vbuf, gbuf, bbuf, obuf);

  // 6) gated rmsnorm * silu(z) -> bf16
  gate_kernel<<<(TOK * NVH) / 8, 256, 0, stream>>>(obuf, qkvz, normk, og);

  // 7) out = og @ W_out  (M=4096, N=2048, K=4096) -> f32 d_out
  wmma_gemm_kernel<false><<<dim3(HID / 256, TOK / 32), 256, 0, stream>>>(
      og, wot, d_out, TOK, HID, VAL_DIM);
}
